// FakeClassifier_29497835389445
// MI455X (gfx1250) — compile-verified, matrix operand order checked by reading, not run on hardware
//
#include <hip/hip_runtime.h>
#include <hip/hip_bf16.h>

// ---------------- problem constants ----------------
#define BB    32768   // batch
#define HH    512     // hidden
#define MD    8       // meta embed dim
#define CD    1024    // classifier dim
#define MV    1000    // meta vocab
#define NCAT  262144  // ragged category ids
#define KIN   532     // H + 20
#define KPAD  544     // KIN padded to multiple of 32
#define KT    17      // KPAD / 32
#define EPSF  1e-6f
#define MTILE 32      // rows per block (2 A fragments per wave)

typedef __attribute__((ext_vector_type(16))) __bf16 v16bf;
typedef __attribute__((ext_vector_type(8)))  float  v8f;

// Self-inverse permutation mapping packed position <-> local K within a 32-wide
// K tile, matching the 16-bit A/B fragment VGPR layout (half0: K 0..7,16..23;
// half1: K 8..15,24..31 -> store order [0..7,16..23,8..15,24..31]).
__device__ __forceinline__ int kperm(int pos) {
    int b = (pos >> 3) & 3;
    return pos + ((b == 1) ? 8 : (b == 2) ? -8 : 0);
}

// ---------------- kernel 1: zero segment scratch ----------------
__global__ void k_zero(float* p, int n) {
    int i = blockIdx.x * blockDim.x + threadIdx.x;
    if (i < n) p[i] = 0.0f;
}

// ---------------- kernel 2: ragged segment sum (sorted seg ids, atomics) ----
__global__ void k_segsum(const int* __restrict__ cat_ids,
                         const int* __restrict__ cat_seg,
                         const float* __restrict__ meta_table,
                         float* __restrict__ sums,     // [B][8]
                         float* __restrict__ counts) { // [B]
    int t = blockIdx.x * blockDim.x + threadIdx.x;
    if (t >= NCAT) return;
    int cid = cat_ids[t];
    int seg = cat_seg[t];
    const float* row = meta_table + (size_t)cid * MD;
#pragma unroll
    for (int d = 0; d < MD; ++d) atomicAdd(&sums[(size_t)seg * MD + d], row[d]);
    atomicAdd(&counts[seg], 1.0f);
}

// ---------------- kernel 3: meta features + credit biases ----------------
__global__ void k_meta(const float* __restrict__ sums,
                       const float* __restrict__ counts,
                       const float* __restrict__ meta_table,
                       const int*   __restrict__ meta_ids,   // [B][5]
                       const float* __restrict__ conv_w,     // [5][6][3]
                       const float* __restrict__ conv_b,     // [5]
                       const float* __restrict__ credit_vec, // [B][6]
                       float* __restrict__ meta_feat,        // [B][20]
                       float* __restrict__ biases) {         // [B][6]
    int i = blockIdx.x * blockDim.x + threadIdx.x;
    if (i >= BB) return;

    float mm[6][MD];                                   // meta_mat row-major
    float inv = 1.0f / fmaxf(counts[i], 1.0f);
#pragma unroll
    for (int d = 0; d < MD; ++d) mm[0][d] = sums[(size_t)i * MD + d] * inv;
#pragma unroll
    for (int r = 0; r < 5; ++r) {
        int id = meta_ids[i * 5 + r];
        const float* row = meta_table + (size_t)id * MD;
#pragma unroll
        for (int d = 0; d < MD; ++d) mm[r + 1][d] = row[d];
    }
    // Conv1d(6->5, k=3, VALID) + ReLU  ->  [5][6]
    float cv[5][6];
#pragma unroll
    for (int c = 0; c < 5; ++c)
#pragma unroll
        for (int x = 0; x < 6; ++x) {
            float a = conv_b[c];
#pragma unroll
            for (int ic = 0; ic < 6; ++ic)
#pragma unroll
                for (int kx = 0; kx < 3; ++kx)
                    a = fmaf(mm[ic][x + kx], conv_w[c * 18 + ic * 3 + kx], a);
            cv[c][x] = fmaxf(a, 0.0f);
        }
    // MaxPool1d(3, stride 1) -> [5][4], flatten
#pragma unroll
    for (int c = 0; c < 5; ++c)
#pragma unroll
        for (int x = 0; x < 4; ++x)
            meta_feat[(size_t)i * 20 + c * 4 + x] =
                fmaxf(cv[c][x], fmaxf(cv[c][x + 1], cv[c][x + 2]));

    // biases: normalized credit vec, uniform fallback
    float s = 0.0f;
#pragma unroll
    for (int j = 0; j < 6; ++j) s += credit_vec[(size_t)i * 6 + j];
#pragma unroll
    for (int j = 0; j < 6; ++j)
        biases[(size_t)i * 6 + j] = (s > 0.0f) ? credit_vec[(size_t)i * 6 + j] / s
                                               : (1.0f / 6.0f);
}

// ---------------- kernel 4: LN1 + bf16 pack in A-fragment order --------------
__device__ __forceinline__ float fetch_in(const float* __restrict__ enc,
                                          const float* __restrict__ mf,
                                          int row, int k) {
    return (k < HH) ? enc[(size_t)row * HH + k] : mf[(size_t)row * 20 + (k - HH)];
}

__global__ void k_ln1pack(const float* __restrict__ encode,
                          const float* __restrict__ meta_feat,
                          const float* __restrict__ ln1_g,
                          const float* __restrict__ ln1_b,
                          __bf16* __restrict__ X) {   // [B][KPAD] packed
    int row = blockIdx.x;
    int tid = threadIdx.x;          // 256 threads
    __shared__ float red[512];

    float s = 0.0f, ss = 0.0f;
    for (int k = tid; k < KIN; k += 256) {
        float v = fetch_in(encode, meta_feat, row, k);
        s += v; ss += v * v;
    }
    red[tid] = s; red[256 + tid] = ss;
    __syncthreads();
    for (int off = 128; off > 0; off >>= 1) {
        if (tid < off) { red[tid] += red[tid + off]; red[256 + tid] += red[256 + tid + off]; }
        __syncthreads();
    }
    float mu   = red[0] * (1.0f / KIN);
    float var  = red[256] * (1.0f / KIN) - mu * mu;
    float rstd = rsqrtf(var + EPSF);

    for (int p = tid; p < KPAD; p += 256) {
        int kt  = p >> 5;
        int k   = kt * 32 + kperm(p & 31);
        float o = 0.0f;
        if (k < KIN) {
            float v = fetch_in(encode, meta_feat, row, k);
            o = ln1_g[k] * (v - mu) * rstd + ln1_b[k];
        }
        X[(size_t)row * KPAD + p] = (__bf16)o;
    }
}

// ---------------- kernel 5: pack mlp1_w^T to bf16 B-fragment order ----------
__global__ void k_wpack(const float* __restrict__ mlp1_w,  // [CD][KIN]
                        __bf16* __restrict__ Wp) {         // [KT][CD][32]
    int idx = blockIdx.x * blockDim.x + threadIdx.x;
    if (idx >= KT * CD * 32) return;
    int pos = idx & 31;
    int n   = (idx >> 5) & (CD - 1);
    int kt  = idx >> 15;            // 32*CD == 1<<15
    int k   = kt * 32 + kperm(pos);
    Wp[idx] = (k < KIN) ? (__bf16)mlp1_w[(size_t)n * KIN + k] : (__bf16)0.0f;
}

// ---------------- kernel 6: fused WMMA GEMM + ReLU + LN2 + head -------------
// Block = 32 rows x full N=1024. 8 waves; wave w owns N columns [w*128, w*128+128).
// Each wave keeps 2 A fragments (rows 0-15 / 16-31 of the block) so every
// B fragment load feeds two WMMAs (halves W traffic from L2).
__global__ void __launch_bounds__(256)
k_gemm_fused(const __bf16* __restrict__ X,       // [B][KPAD] packed
             const __bf16* __restrict__ Wp,      // [KT][CD][32] packed
             const float* __restrict__ mlp1_b,   // [CD]
             const float* __restrict__ ln2_g,
             const float* __restrict__ ln2_b,
             const float* __restrict__ out_w,    // [6][CD]
             const float* __restrict__ out_b,    // [6]
             const float* __restrict__ biases,   // [B][6]
             float* __restrict__ out) {          // [B][6]
    __shared__ float rsum[MTILE];
    __shared__ float rsq[MTILE];
    __shared__ float rout[MTILE * 6];

    const int tid  = threadIdx.x;
    const int lane = tid & 31;
    const int wave = tid >> 5;
    const int half = lane >> 4;
    const int lm   = lane & 15;
    const int row0 = blockIdx.x * MTILE;

    if (tid < MTILE) { rsum[tid] = 0.0f; rsq[tid] = 0.0f; }
    if (tid < MTILE * 6) { rout[tid] = 0.0f; }

    v8f acc[2][8] = {};
    const __bf16* xrow0 = X + (size_t)(row0 + lm) * KPAD + half * 16;
    const __bf16* xrow1 = xrow0 + (size_t)16 * KPAD;
    const __bf16* wbase = Wp + ((size_t)(wave * 128 + lm)) * 32 + half * 16;

    for (int kt = 0; kt < KT; ++kt) {
        v16bf a0 = *(const v16bf*)(xrow0 + kt * 32);
        v16bf a1 = *(const v16bf*)(xrow1 + kt * 32);
        if (kt + 1 < KT) {
            __builtin_prefetch(xrow0 + (kt + 1) * 32, 0, 0);
            __builtin_prefetch(xrow1 + (kt + 1) * 32, 0, 0);
        }
        const __bf16* wk = wbase + (size_t)kt * CD * 32;
#pragma unroll
        for (int t = 0; t < 8; ++t) {
            v16bf b = *(const v16bf*)(wk + (size_t)t * 16 * 32);
            acc[0][t] = __builtin_amdgcn_wmma_f32_16x16x32_bf16(
                false, a0, false, b, (short)0, acc[0][t], false, false);
            acc[1][t] = __builtin_amdgcn_wmma_f32_16x16x32_bf16(
                false, a1, false, b, (short)0, acc[1][t], false, false);
        }
    }
    __syncthreads();   // shared-init complete; accumulators final

    // bias + ReLU in registers; preload per-column head weights (shared by
    // both row fragments)
    const int nbase = wave * 128 + lm;
    float wg[8], wb2[8], wo[8][6];
#pragma unroll
    for (int t = 0; t < 8; ++t) {
        int n = nbase + t * 16;
        float bc = mlp1_b[n];
#pragma unroll
        for (int f = 0; f < 2; ++f)
#pragma unroll
            for (int r = 0; r < 8; ++r) {
                float v = acc[f][t][r] + bc;
                acc[f][t][r] = v > 0.0f ? v : 0.0f;
            }
        wg[t] = ln2_g[n]; wb2[t] = ln2_b[n];
#pragma unroll
        for (int j = 0; j < 6; ++j) wo[t][j] = out_w[j * CD + n];
    }

    // LN2 row statistics across the block (ds_add_f32)
#pragma unroll
    for (int f = 0; f < 2; ++f)
#pragma unroll
        for (int r = 0; r < 8; ++r) {
            int m = f * 16 + r + 8 * half;
            float s = 0.0f, ss = 0.0f;
#pragma unroll
            for (int t = 0; t < 8; ++t) { float v = acc[f][t][r]; s += v; ss += v * v; }
            atomicAdd(&rsum[m], s);
            atomicAdd(&rsq[m], ss);
        }
    __syncthreads();

    // normalize + project to 6 outputs (partial dot per lane, reduce in LDS)
#pragma unroll
    for (int f = 0; f < 2; ++f)
#pragma unroll
        for (int r = 0; r < 8; ++r) {
            int m = f * 16 + r + 8 * half;
            float mu   = rsum[m] * (1.0f / CD);
            float var  = rsq[m] * (1.0f / CD) - mu * mu;
            float rstd = rsqrtf(var + EPSF);
            float o[6] = {0, 0, 0, 0, 0, 0};
#pragma unroll
            for (int t = 0; t < 8; ++t) {
                float v = (acc[f][t][r] - mu) * rstd * wg[t] + wb2[t];
#pragma unroll
                for (int j = 0; j < 6; ++j) o[j] = fmaf(v, wo[t][j], o[j]);
            }
#pragma unroll
            for (int j = 0; j < 6; ++j) atomicAdd(&rout[m * 6 + j], o[j]);
        }
    __syncthreads();

    if (tid < MTILE * 6) {
        int m = tid / 6, j = tid % 6;
        int row = row0 + m;
        out[(size_t)row * 6 + j] = rout[tid] + out_b[j] + biases[(size_t)row * 6 + j];
    }
}

// ---------------- host-side launch ----------------
extern "C" void kernel_launch(void* const* d_in, const int* in_sizes, int n_in,
                              void* d_out, int out_size, void* d_ws, size_t ws_size,
                              hipStream_t stream) {
    const float* encode     = (const float*)d_in[0];
    const float* credit_vec = (const float*)d_in[1];
    const float* meta_table = (const float*)d_in[2];
    const float* conv_w     = (const float*)d_in[3];
    const float* conv_b     = (const float*)d_in[4];
    const float* ln1_g      = (const float*)d_in[5];
    const float* ln1_b      = (const float*)d_in[6];
    const float* mlp1_w     = (const float*)d_in[7];
    const float* mlp1_b     = (const float*)d_in[8];
    const float* ln2_g      = (const float*)d_in[9];
    const float* ln2_b      = (const float*)d_in[10];
    const float* out_w      = (const float*)d_in[11];
    const float* out_b      = (const float*)d_in[12];
    const int*   meta_ids   = (const int*)d_in[13];
    const int*   cat_ids    = (const int*)d_in[14];
    const int*   cat_seg    = (const int*)d_in[15];
    float*       out        = (float*)d_out;

    // workspace layout (all offsets 64B-aligned)
    char* ws = (char*)d_ws;
    float*  sums      = (float*)ws;                               // B*8
    float*  counts    = sums + (size_t)BB * 8;                    // B
    float*  meta_feat = counts + BB;                              // B*20
    float*  biases    = meta_feat + (size_t)BB * 20;              // B*6
    __bf16* X         = (__bf16*)(biases + (size_t)BB * 6);       // B*KPAD bf16
    __bf16* Wp        = X + (size_t)BB * KPAD;                    // KT*CD*32 bf16

    const int nz = BB * 9;  // sums + counts
    k_zero<<<(nz + 255) / 256, 256, 0, stream>>>(sums, nz);
    k_segsum<<<NCAT / 256, 256, 0, stream>>>(cat_ids, cat_seg, meta_table, sums, counts);
    k_meta<<<BB / 256, 256, 0, stream>>>(sums, counts, meta_table, meta_ids,
                                         conv_w, conv_b, credit_vec,
                                         meta_feat, biases);
    k_ln1pack<<<BB, 256, 0, stream>>>(encode, meta_feat, ln1_g, ln1_b, X);
    k_wpack<<<(KT * CD * 32 + 255) / 256, 256, 0, stream>>>(mlp1_w, Wp);
    k_gemm_fused<<<BB / MTILE, 256, 0, stream>>>(X, Wp, mlp1_b, ln2_g, ln2_b,
                                                 out_w, out_b, biases, out);
}
